// DeformConvWithOffset_51092930953234
// MI455X (gfx1250) — matrix operand order repaired and checked by reading, not compiled, not run on hardware
//
#include <hip/hip_runtime.h>
#include <stdint.h>

typedef __attribute__((ext_vector_type(16))) _Float16     v16h;
typedef __attribute__((ext_vector_type(8)))  float        v8f;
typedef __attribute__((ext_vector_type(4)))  float        f4;
typedef __attribute__((ext_vector_type(4)))  int          i32x4;
typedef __attribute__((ext_vector_type(8)))  int          i32x8;
typedef __attribute__((ext_vector_type(4)))  unsigned int u32x4;

#define Bq   4
#define Cc   256
#define Hh   96
#define Ww   96
#define Oo   256
#define HW   (Hh*Ww)      // 9216
#define Kdim (Cc*9)       // 2304
#define Mtot (Bq*HW)      // 36864

// ---------------- Stage 1: offset-predicting 3x3 conv (pad 1) ----------------
__global__ __launch_bounds__(256) void offset_conv(
    const float* __restrict__ x, const float* __restrict__ w_off,
    const float* __restrict__ b_off, float* __restrict__ off) {
  int idx = blockIdx.x * 256 + threadIdx.x;        // total = 4*18*96*96
  int xc = idx % Ww;
  int y  = (idx / Ww) % Hh;
  int oc = (idx / HW) % 18;                        // uniform per block
  int b  = idx / (HW * 18);
  float acc = b_off[oc];
  const float* xb = x + b * Cc * HW;
  const float* wo = w_off + oc * Cc * 9;
#pragma unroll
  for (int ky = 0; ky < 3; ++ky) {
    int iy = y - 1 + ky;
    if (iy < 0 || iy >= Hh) continue;
#pragma unroll
    for (int kx = 0; kx < 3; ++kx) {
      int ix = xc - 1 + kx;
      if (ix < 0 || ix >= Ww) continue;
      const float* xp = xb + iy * Ww + ix;
      const float* wp = wo + ky * 3 + kx;
      for (int c = 0; c < Cc; ++c)
        acc = fmaf(xp[c * HW], wp[c * 9], acc);
    }
  }
  off[idx] = acc;
}

// ------------- Stage 1b: convert w_dc to f16 once ([n][k] layout) ------------
__global__ __launch_bounds__(256) void wcvt(
    const float* __restrict__ w, _Float16* __restrict__ o) {
  int i = blockIdx.x * 256 + threadIdx.x;          // total = 256*2304
  o[i] = (_Float16)w[i];
}

// ---- TDM: 2D tile (tile_dim0=32 k, tile_dim1=256 n) of f16 weights -> LDS ----
__device__ __forceinline__ void tdm_load_b_tile(const _Float16* gsrc,
                                                unsigned lds_off) {
  unsigned long long ga = (unsigned long long)(uintptr_t)gsrc;
  u32x4 g0 = { 1u,                                  // count=1, user descriptor
               lds_off,                             // LDS byte address
               (unsigned)ga,                        // global_addr[31:0]
               (unsigned)((ga >> 32) & 0x1FFFFFFull) | (2u << 30) }; // type=2
  i32x8 g1 = { 0x10000,                             // data_size=1 (2 bytes)
               (int)(Kdim << 16),                   // tensor_dim0 = 2304
               (int)(Oo << 16),                     // tensor_dim1 = 256
               (int)(32 << 16),                     // tile_dim0   = 32
               Oo,                                  // tile_dim1   = 256
               Kdim,                                // dim0_stride = 2304
               0, 0 };
  i32x4 gz4 = { 0, 0, 0, 0 };
  i32x8 gz8 = { 0, 0, 0, 0, 0, 0, 0, 0 };
  __builtin_amdgcn_tensor_load_to_lds(g0, g1, gz4, gz4, gz8, 0);
}

// ------------- Stage 2: fused bilinear-sample + implicit GEMM (WMMA) ---------
// out[M=36864, N=256] = A[M, K=2304] x B[K, N],  K index = c*9 + tap
// Block tile: 32(M) x 256(N), 8 waves, 4 WMMA tiles each; B double-buffered TDM.
__global__ __launch_bounds__(256, 2) void deform_gemm(
    const float* __restrict__ x, const float* __restrict__ off,
    const _Float16* __restrict__ wB, const float* __restrict__ b_dc,
    float* __restrict__ out) {
  __shared__ __align__(16) int sP[288 * 8];   // per (m,tap): 4 idx + 4 wgt
  __shared__ _Float16 lA[32 * 32];            // A tile [m][k]
  __shared__ _Float16 lB[2][256 * 32];        // B tile [n][k], double buffered

  const int tid  = threadIdx.x;
  const int m0   = blockIdx.x * 32;
  const int b    = m0 / HW;                   // uniform (9216 % 32 == 0)
  const int pos0 = m0 % HW;
  const int y    = pos0 / Ww;                 // uniform (96 % 32 == 0)
  const int x0   = pos0 % Ww;

  // --- precompute bilinear sampling params: 32 rows x 9 taps ---
  for (int e = tid; e < 288; e += 256) {
    int ml = e / 9, t = e % 9;
    int xc = x0 + ml;
    float dy = off[((b * 18 + t * 2 + 0) * Hh + y) * Ww + xc];
    float dx = off[((b * 18 + t * 2 + 1) * Hh + y) * Ww + xc];
    float py = (float)(y  - 1 + t / 3) + dy;
    float px = (float)(xc - 1 + t % 3) + dx;
    float fy = floorf(py), fx = floorf(px);
    int y0i = (int)fy, x0i = (int)fx;
    float ly = py - fy, lx = px - fx;
    float w00 = (1.f - ly) * (1.f - lx), w01 = (1.f - ly) * lx;
    float w10 = ly * (1.f - lx),         w11 = ly * lx;
    int y1i = y0i + 1, x1i = x0i + 1;
    bool vy0 = (y0i >= 0) & (y0i < Hh), vy1 = (y1i >= 0) & (y1i < Hh);
    bool vx0 = (x0i >= 0) & (x0i < Ww), vx1 = (x1i >= 0) & (x1i < Ww);
    int cy0 = min(max(y0i, 0), Hh - 1), cy1 = min(max(y1i, 0), Hh - 1);
    int cx0 = min(max(x0i, 0), Ww - 1), cx1 = min(max(x1i, 0), Ww - 1);
    int p8 = e * 8;
    sP[p8 + 0] = cy0 * Ww + cx0;  sP[p8 + 1] = cy0 * Ww + cx1;
    sP[p8 + 2] = cy1 * Ww + cx0;  sP[p8 + 3] = cy1 * Ww + cx1;
    float* wq = (float*)sP + p8 + 4;
    wq[0] = (vy0 && vx0) ? w00 : 0.f;  wq[1] = (vy0 && vx1) ? w01 : 0.f;
    wq[2] = (vy1 && vx0) ? w10 : 0.f;  wq[3] = (vy1 && vx1) ? w11 : 0.f;
  }

  const int wave = tid >> 5;
  const int lane = tid & 31;
  const int half = lane >> 4;
  const int lrow = lane & 15;
  const int mt   = wave & 1;                  // M subtile (0..1)
  const int ntb  = (wave >> 1) * 4;           // N subtile base (0,4,8,12)

  v8f acc[4] = {};
  const float* xb = x + b * Cc * HW;

  // prologue: TDM tile 0 into buffer 0
  if (wave == 0)
    tdm_load_b_tile(wB + 0, (unsigned)(uintptr_t)&lB[0][0]);

  for (int it = 0, k0 = 0; it < Kdim / 32; ++it, k0 += 32) {
    __syncthreads();                          // lA readers of prev iter done
    // ---- fill A tile: 32x32, bilinear-sampled, f32 -> f16 ----
#pragma unroll
    for (int i = 0; i < 4; ++i) {
      int e  = tid + 256 * i;
      int ml = e & 31, kk = e >> 5;
      int k  = k0 + kk;
      int c  = k / 9, t = k % 9;
      const float* xp = xb + c * HW;
      int p8 = (ml * 9 + t) * 8;
      i32x4 id = *(const i32x4*)(sP + p8);              // ds_load_b128
      f4    wq = *(const f4*)((const float*)sP + p8 + 4);
      float v = wq.x * xp[id.x] + wq.y * xp[id.y]
              + wq.z * xp[id.z] + wq.w * xp[id.w];
      lA[ml * 32 + kk] = (_Float16)v;
    }
    // ---- prefetch next B tile via TDM, ensure current tile landed ----
    if (wave == 0) {
      int kn = k0 + 32;
      if (kn >= Kdim) kn = 0;                 // dummy wrap on last iter
      tdm_load_b_tile(wB + kn, (unsigned)(uintptr_t)&lB[(it + 1) & 1][0]);
      __builtin_amdgcn_s_wait_tensorcnt(1);   // current tile complete
    }
    __syncthreads();

    const _Float16* lBc = &lB[it & 1][0];

    // ---- A fragment (ISA 16-bit A 16x32 layout) ----
    const _Float16* ap = lA + (mt * 16 + lrow) * 32;
    v16h afrag;
#pragma unroll
    for (int v = 0; v < 8; ++v) {
      int kb = (v < 4) ? (2 * v + 8 * half) : (16 + 2 * (v - 4) + 8 * half);
      afrag[2 * v]     = ap[kb];
      afrag[2 * v + 1] = ap[kb + 1];
    }
    // ---- 4 N-subtiles per wave ----
#pragma unroll
    for (int j = 0; j < 4; ++j) {
      const _Float16* bp = lBc + ((ntb + j) * 16 + lrow) * 32;
      v16h bfrag;
#pragma unroll
      for (int v = 0; v < 8; ++v) {
        int kb = (v < 4) ? (2 * v + 8 * half) : (16 + 2 * (v - 4) + 8 * half);
        bfrag[2 * v]     = bp[kb];
        bfrag[2 * v + 1] = bp[kb + 1];
      }
      acc[j] = __builtin_amdgcn_wmma_f32_16x16x32_f16(
          false, afrag, false, bfrag, (short)0, acc[j], false, false);
    }
  }

  // ---- epilogue: D layout VGPR v -> row M = 8*half + v, col N = lane%16 ----
#pragma unroll
  for (int j = 0; j < 4; ++j) {
    int o = (ntb + j) * 16 + lrow;
    float bias = b_dc[o];
    float* op = out + (size_t)(b * Oo + o) * HW;
#pragma unroll
    for (int v = 0; v < 8; ++v) {
      int Mrow = mt * 16 + 8 * half + v;
      op[pos0 + Mrow] = acc[j][v] + bias;     // same W-row: x0 % 32 == 0
    }
  }
}

extern "C" void kernel_launch(void* const* d_in, const int* in_sizes, int n_in,
                              void* d_out, int out_size, void* d_ws, size_t ws_size,
                              hipStream_t stream) {
  const float* x     = (const float*)d_in[0];
  const float* w_off = (const float*)d_in[1];
  const float* b_off = (const float*)d_in[2];
  const float* w_dc  = (const float*)d_in[3];
  const float* b_dc  = (const float*)d_in[4];
  float* out = (float*)d_out;

  float*    off = (float*)d_ws;                       // 663552 f32 = 2.65 MB
  _Float16* wB  = (_Float16*)((char*)d_ws + (size_t)Bq * 18 * HW * 4); // 1.18 MB

  int n_off = Bq * 18 * HW;                           // 663552
  offset_conv<<<n_off / 256, 256, 0, stream>>>(x, w_off, b_off, off);
  wcvt<<<(Oo * Kdim) / 256, 256, 0, stream>>>(w_dc, wB);
  deform_gemm<<<Mtot / 32, 256, 0, stream>>>(x, off, wB, b_dc, out);
}
